// TargetRollout_26663156973627
// MI455X (gfx1250) — compile-verified, matrix operand order checked by reading, not run on hardware
//
#include <hip/hip_runtime.h>
#include <math.h>

// ---------------------------------------------------------------------------
// TargetRollout for MI455X (gfx1250): WMMA f32_16x16x32_f16 pipeline, v3.
// v3: GRU step uses 4-wave blocks with all weights staged into LDS via
// CDNA5 async global->LDS loads (ASYNCcnt), 4x less L2 weight traffic.
// B=32, N_OBS=N_TAR=512, HDIM=128, NHEADS=4, HEAD_DIM=32, M=250(+pad->256), H=24.
// ---------------------------------------------------------------------------

typedef __attribute__((ext_vector_type(16))) _Float16 v16h;
typedef __attribute__((ext_vector_type(8)))  _Float16 v8h;
typedef __attribute__((ext_vector_type(8)))  float    v8f;

#define HDIM   128
#define NTAR   512
#define NOBS   512
#define BATCH  32
#define NHEADS 4
#define HEADD  32
#define HSTEPS 24
#define MDIM   250
#define MPAD   256

static __device__ __forceinline__ int lane_id() { return (int)(threadIdx.x & 31); }

// ---------------------------------------------------------------------------
// Fragment loaders. 16-bit 16x32 fragment layout (ISA 7.12.2): per lane,
// elements 0..7 are K = k0+kh+0..7 and elements 8..15 are K = k0+16+kh+0..7,
// kh = 8 for lanes 16..31; row/col = lane & 15.  Both runs are contiguous,
// so an f16 row-major source needs exactly two 16-byte loads.
// ---------------------------------------------------------------------------
static __device__ __forceinline__ v16h load_frag_f16(const _Float16* S, int row0,
                                                     int lda, int k0) {
  int lane = lane_id();
  const _Float16* p =
      S + (size_t)(row0 + (lane & 15)) * lda + k0 + ((lane >= 16) ? 8 : 0);
  v8h lo = *(const v8h*)(p);
  v8h hi = *(const v8h*)(p + 16);
  return __builtin_shufflevector(lo, hi, 0, 1, 2, 3, 4, 5, 6, 7, 8, 9, 10, 11, 12,
                                 13, 14, 15);
}

static __device__ __forceinline__ v16h load_frag_f32(const float* S, int row0,
                                                     int lda, int k0) {
  int lane = lane_id();
  const float* p =
      S + (size_t)(row0 + (lane & 15)) * lda + k0 + ((lane >= 16) ? 8 : 0);
  float4 a0 = ((const float4*)p)[0];
  float4 a1 = ((const float4*)p)[1];
  float4 b0 = ((const float4*)(p + 16))[0];
  float4 b1 = ((const float4*)(p + 16))[1];
  v16h f;
  f[0] = (_Float16)a0.x;  f[1] = (_Float16)a0.y;  f[2] = (_Float16)a0.z;  f[3] = (_Float16)a0.w;
  f[4] = (_Float16)a1.x;  f[5] = (_Float16)a1.y;  f[6] = (_Float16)a1.z;  f[7] = (_Float16)a1.w;
  f[8] = (_Float16)b0.x;  f[9] = (_Float16)b0.y;  f[10] = (_Float16)b0.z; f[11] = (_Float16)b0.w;
  f[12] = (_Float16)b1.x; f[13] = (_Float16)b1.y; f[14] = (_Float16)b1.z; f[15] = (_Float16)b1.w;
  return f;
}

static __device__ __forceinline__ v8f wmma_f16(v16h a, v16h b, v8f c) {
  return __builtin_amdgcn_wmma_f32_16x16x32_f16(false, a, false, b, (short)0, c,
                                                false, false);
}

// C/D layout: VGPR r holds M = r + 8*(lane>=16), N = lane&15.
static __device__ __forceinline__ void store_c_f32(float* C, int row0, int ldc,
                                                   int col0, v8f c) {
  int lane = lane_id();
  int n = col0 + (lane & 15);
  int mb = row0 + ((lane >= 16) ? 8 : 0);
#pragma unroll
  for (int r = 0; r < 8; ++r) C[(size_t)(mb + r) * ldc + n] = c[r];
}

static __device__ __forceinline__ void store_c_f16(_Float16* C, int row0, int ldc,
                                                   int col0, v8f c) {
  int lane = lane_id();
  int n = col0 + (lane & 15);
  int mb = row0 + ((lane >= 16) ? 8 : 0);
#pragma unroll
  for (int r = 0; r < 8; ++r) C[(size_t)(mb + r) * ldc + n] = (_Float16)c[r];
}

// ---------------------------------------------------------------------------
// CDNA5 async global->LDS copy (ISA 15.18, GLOBAL_LOAD_ASYNC_TO_LDS_B128,
// tracked by ASYNCcnt).  VDST = LDS byte address = low 32 bits of the generic
// LDS pointer; 16 bytes per issued op per lane.
// ---------------------------------------------------------------------------
static __device__ __forceinline__ void async_copy_f16(const _Float16* __restrict__ g,
                                                      unsigned int lds_byte,
                                                      int halves, int tid,
                                                      int nthreads) {
  for (int i = tid * 8; i < halves; i += nthreads * 8) {
    unsigned int dst = lds_byte + (unsigned int)(i * 2);
    unsigned long long src = (unsigned long long)(size_t)(g + i);
    asm volatile("global_load_async_to_lds_b128 %0, %1, off" ::"v"(dst), "v"(src)
                 : "memory");
  }
}

static __device__ __forceinline__ void async_wait0() {
  asm volatile("s_wait_asynccnt 0x0" ::: "memory");
}

// ---------------------------------------------------------------------------
// f32 -> f16 staging kernels
// ---------------------------------------------------------------------------
__global__ void cvt_f16(const float* __restrict__ src, _Float16* __restrict__ dst,
                        int n) {
  int i = blockIdx.x * 256 + threadIdx.x;
  if (i < n) dst[i] = (_Float16)src[i];
}

__global__ void pad_cvt_f16(const float* __restrict__ src, _Float16* __restrict__ dst,
                            int rows, int cs, int cd) {
  int i = blockIdx.x * 256 + threadIdx.x;
  if (i >= rows * cd) return;
  int r = i / cd, c = i % cd;
  dst[i] = (c < cs) ? (_Float16)src[(size_t)r * cs + c] : (_Float16)0.0f;
}

// ---------------------------------------------------------------------------
// Y[m,n] = sum_k X[m,k]*W[n,k] + bias[n].  One wave -> 16x64 tile (A reuse x4).
// OUT_MODE 0: f32 row-major; 1: f16 row-major; 2: f16 transposed per-batch
// (for V^T: dst[((m0/512)*HDIM + n)*NTAR + m], contiguous 16B store per lane).
// ---------------------------------------------------------------------------
template <int OUT_MODE>
__global__ void gemm_f16(const _Float16* __restrict__ X, int ldx,
                         const _Float16* __restrict__ W, int ldw,
                         const float* __restrict__ bias, void* __restrict__ Y,
                         int ldy, int K) {
  int m0 = blockIdx.x * 16;
  int n0 = blockIdx.y * 64;
  int lane = lane_id();
  v8f acc[4] = {{}, {}, {}, {}};
  for (int k0 = 0; k0 < K; k0 += 32) {
    v16h a = load_frag_f16(X, m0, ldx, k0);
#pragma unroll
    for (int t = 0; t < 4; ++t) {
      v16h b = load_frag_f16(W, n0 + t * 16, ldw, k0);
      acc[t] = wmma_f16(a, b, acc[t]);
    }
  }
#pragma unroll
  for (int t = 0; t < 4; ++t) {
    int nt = n0 + t * 16;
    float bb = bias[nt + (lane & 15)];
#pragma unroll
    for (int r = 0; r < 8; ++r) acc[t][r] += bb;
    if (OUT_MODE == 0) {
      store_c_f32((float*)Y, m0, ldy, nt, acc[t]);
    } else if (OUT_MODE == 1) {
      store_c_f16((_Float16*)Y, m0, ldy, nt, acc[t]);
    } else {
      int b = m0 >> 9;
      int mloc = (m0 & 511) + ((lane >= 16) ? 8 : 0);
      _Float16* dst = (_Float16*)Y +
                      ((size_t)b * HDIM + nt + (lane & 15)) * NTAR + mloc;
      v8h o;
#pragma unroll
      for (int r = 0; r < 8; ++r) o[r] = (_Float16)acc[t][r];
      *(v8h*)dst = o;
    }
  }
}

// ---------------------------------------------------------------------------
// Fused attention: per (b, 16-row t-tile), all 4 heads.
// scores (1 WMMA per 16x16, K=32) -> float4 softmax (all 32 lanes, shfl_xor
// cross-half reduce) -> head-mean to d_out (float4) -> w@v (WMMA, V^T f16).
// ---------------------------------------------------------------------------
__global__ void attn_kernel(const _Float16* __restrict__ qbuf,  // (NTAR,HDIM)
                            const _Float16* __restrict__ kbuf,  // (B*NOBS,HDIM)
                            const _Float16* __restrict__ vT,    // (B,HDIM,NOBS)
                            _Float16* __restrict__ attn_out,    // (B*NTAR,HDIM)
                            float* __restrict__ attnW) {        // (B,NTAR,NOBS)
  __shared__ __align__(16) float sw[16 * NOBS];  // 32 KB
  int t0 = blockIdx.x * 16;
  int b = blockIdx.y;
  int lane = lane_id();
  const _Float16* kb = kbuf + (size_t)b * NOBS * HDIM;
  const _Float16* vb = vT + (size_t)b * HDIM * NOBS;
  const float scale = 0.17677669529663687f;  // 1/sqrt(32)

  for (int h = 0; h < NHEADS; ++h) {
    int c0 = h * HEADD;
    v16h aq = load_frag_f16(qbuf + c0, t0, HDIM, 0);
    for (int s0 = 0; s0 < NOBS; s0 += 16) {
      v16h bk = load_frag_f16(kb + c0, s0, HDIM, 0);
      v8f c = {};
      c = wmma_f16(aq, bk, c);
      int n = s0 + (lane & 15);
      int mb = (lane >= 16) ? 8 : 0;
#pragma unroll
      for (int r = 0; r < 8; ++r) sw[(size_t)(mb + r) * NOBS + n] = c[r] * scale;
    }
    __syncthreads();

    // softmax: lanes l and l+16 co-own row l&15, each scans 64 float4's.
    {
      int r = lane & 15;
      int half = lane >> 4;
      float4* row4 = (float4*)(sw + (size_t)r * NOBS) + half * 64;
      float mx = -1e30f;
      for (int i = 0; i < 64; ++i) {
        float4 v = row4[i];
        mx = fmaxf(mx, fmaxf(fmaxf(v.x, v.y), fmaxf(v.z, v.w)));
      }
      mx = fmaxf(mx, __shfl_xor(mx, 16, 32));
      float sum = 0.0f;
      for (int i = 0; i < 64; ++i) {
        float4 v = row4[i];
        v.x = __expf(v.x - mx); v.y = __expf(v.y - mx);
        v.z = __expf(v.z - mx); v.w = __expf(v.w - mx);
        row4[i] = v;
        sum += (v.x + v.y) + (v.z + v.w);
      }
      sum += __shfl_xor(sum, 16, 32);
      float inv = 1.0f / sum;
      float4* aw4 =
          (float4*)(attnW + ((size_t)b * NTAR + t0 + r) * NOBS) + half * 64;
      if (h == 0) {
        for (int i = 0; i < 64; ++i) {
          float4 v = row4[i];
          v.x *= inv; v.y *= inv; v.z *= inv; v.w *= inv;
          row4[i] = v;
          float4 o; o.x = 0.25f * v.x; o.y = 0.25f * v.y;
          o.z = 0.25f * v.z; o.w = 0.25f * v.w;
          aw4[i] = o;
        }
      } else {
        for (int i = 0; i < 64; ++i) {
          float4 v = row4[i];
          v.x *= inv; v.y *= inv; v.z *= inv; v.w *= inv;
          row4[i] = v;
          float4 o = aw4[i];
          o.x += 0.25f * v.x; o.y += 0.25f * v.y;
          o.z += 0.25f * v.z; o.w += 0.25f * v.w;
          aw4[i] = o;
        }
      }
    }
    __syncthreads();

    // out[t, c0+n] = sum_s w[t,s] * v[s, c0+n]   (K = 512, B = V^T rows)
#pragma unroll
    for (int nn = 0; nn < 2; ++nn) {
      v8f acc = {};
      for (int s0 = 0; s0 < NOBS; s0 += 32) {
        v16h awf = load_frag_f32(sw, 0, NOBS, s0);
        v16h bv = load_frag_f16(vb + (size_t)c0 * NOBS, nn * 16, NOBS, s0);
        acc = wmma_f16(awf, bv, acc);
      }
      store_c_f16(attn_out, b * NTAR + t0, HDIM, c0 + nn * 16, acc);
    }
    __syncthreads();
  }
}

// ---------------------------------------------------------------------------
// Fused GRU step, v3: 4-wave block owns 64 rows.  All three weight matrices
// (224 KB f16) staged once per block into dynamic LDS via async global->LDS
// loads; every weight fragment then comes from ds_load_b128.  f16 state
// shadow feeds fragments; f32 master state feeds the z*h term.
// LDS: Wh2h[16384] | Wih[49152] | Whh[49152] | 4 x-tiles[4*2048]  (240 KB)
// ---------------------------------------------------------------------------
__global__ void gru_step(const float* __restrict__ h32_in,
                         const _Float16* __restrict__ h16_in,
                         float* __restrict__ h32_out, _Float16* __restrict__ h16_out,
                         const _Float16* __restrict__ Wh2h, const float* __restrict__ b_h2h,
                         const _Float16* __restrict__ Wih, const float* __restrict__ b_ih,
                         const _Float16* __restrict__ Whh, const float* __restrict__ b_hh,
                         float* __restrict__ A, int step) {
  extern __shared__ __align__(16) _Float16 smem[];
  _Float16* sWh2h = smem;            // 16384 halves
  _Float16* sWih  = smem + 16384;    // 49152 halves
  _Float16* sWhh  = smem + 65536;    // 49152 halves
  int tid = (int)threadIdx.x;        // 0..127
  int wave = tid >> 5;
  int lane = tid & 31;
  _Float16* sx = smem + 114688 + wave * (16 * HDIM);

  // stage weights: async global->LDS, then wait ASYNCcnt and barrier
  unsigned int lds0 = (unsigned int)(size_t)smem;  // LDS byte addr = addr[31:0]
  async_copy_f16(Wh2h, lds0, 16384, tid, 128);
  async_copy_f16(Wih, lds0 + 32768u, 49152, tid, 128);
  async_copy_f16(Whh, lds0 + 131072u, 49152, tid, 128);
  async_wait0();
  __syncthreads();

  int row0 = (int)blockIdx.x * 64 + wave * 16;

  v16h ah[4];
#pragma unroll
  for (int kf = 0; kf < 4; ++kf) ah[kf] = load_frag_f16(h16_in, row0, HDIM, kf * 32);

  // x = h @ W_h2h^T + b_h2h  -> LDS (f16), wave-private tile
  for (int n0 = 0; n0 < HDIM; n0 += 16) {
    v8f acc = {};
#pragma unroll
    for (int kf = 0; kf < 4; ++kf) {
      v16h bw = load_frag_f16(sWh2h, n0, HDIM, kf * 32);
      acc = wmma_f16(ah[kf], bw, acc);
    }
    float bb = b_h2h[n0 + (lane & 15)];
    int n = n0 + (lane & 15);
    int mb = (lane >= 16) ? 8 : 0;
#pragma unroll
    for (int r = 0; r < 8; ++r)
      sx[(size_t)(mb + r) * HDIM + n] = (_Float16)(acc[r] + bb);
  }
  __syncthreads();

  v16h ax[4];
#pragma unroll
  for (int kf = 0; kf < 4; ++kf) ax[kf] = load_frag_f16(sx, 0, HDIM, kf * 32);

  for (int n0 = 0; n0 < HDIM; n0 += 16) {
    v8f ir = {}, iz = {}, inn = {}, hr = {}, hz = {}, hn = {};
#pragma unroll
    for (int kf = 0; kf < 4; ++kf) {
      v16h w;
      w = load_frag_f16(sWih, n0,            HDIM, kf * 32); ir  = wmma_f16(ax[kf], w, ir);
      w = load_frag_f16(sWih, n0 + HDIM,     HDIM, kf * 32); iz  = wmma_f16(ax[kf], w, iz);
      w = load_frag_f16(sWih, n0 + 2 * HDIM, HDIM, kf * 32); inn = wmma_f16(ax[kf], w, inn);
      w = load_frag_f16(sWhh, n0,            HDIM, kf * 32); hr  = wmma_f16(ah[kf], w, hr);
      w = load_frag_f16(sWhh, n0 + HDIM,     HDIM, kf * 32); hz  = wmma_f16(ah[kf], w, hz);
      w = load_frag_f16(sWhh, n0 + 2 * HDIM, HDIM, kf * 32); hn  = wmma_f16(ah[kf], w, hn);
    }
    int n = n0 + (lane & 15);
    float bir = b_ih[n], biz = b_ih[n + HDIM], bin = b_ih[n + 2 * HDIM];
    float bhr = b_hh[n], bhz = b_hh[n + HDIM], bhn = b_hh[n + 2 * HDIM];
    int mb = row0 + ((lane >= 16) ? 8 : 0);
#pragma unroll
    for (int r = 0; r < 8; ++r) {
      int mrow = mb + r;
      float hcur = h32_in[(size_t)mrow * HDIM + n];
      float rg = 1.0f / (1.0f + __expf(-(ir[r] + bir + hr[r] + bhr)));
      float zg = 1.0f / (1.0f + __expf(-(iz[r] + biz + hz[r] + bhz)));
      float ng = tanhf(inn[r] + bin + rg * (hn[r] + bhn));
      float hnew = (1.0f - zg) * ng + zg * hcur;
      h32_out[(size_t)mrow * HDIM + n] = hnew;
      h16_out[(size_t)mrow * HDIM + n] = (_Float16)hnew;
      int bidx = mrow >> 9;
      int tr = mrow & 511;
      A[(((size_t)bidx * HSTEPS + step) * NTAR + tr) * HDIM + n] = hnew;
    }
  }
}

// h0 (f32) -> A[:,0] slice + f16 state shadow.
__global__ void seed_state(const float* __restrict__ h0, float* __restrict__ A,
                           _Float16* __restrict__ h16) {
  size_t idx = (size_t)blockIdx.x * blockDim.x + threadIdx.x;
  if (idx >= (size_t)BATCH * NTAR * HDIM) return;
  float v = h0[idx];
  h16[idx] = (_Float16)v;
  int c = (int)(idx % HDIM);
  int m = (int)(idx / HDIM);
  int b = m >> 9, t = m & 511;
  A[(((size_t)b * HSTEPS) * NTAR + t) * HDIM + c] = v;
}

// ---------------------------------------------------------------------------
extern "C" void kernel_launch(void* const* d_in, const int* in_sizes, int n_in,
                              void* d_out, int out_size, void* d_ws, size_t ws_size,
                              hipStream_t stream) {
  const float* H_obs = (const float*)d_in[0];
  const float* phi_t = (const float*)d_in[1];
  const float* W_phi = (const float*)d_in[2];
  const float* b_phi = (const float*)d_in[3];
  const float* W_in  = (const float*)d_in[4];
  const float* b_in  = (const float*)d_in[5];
  const float* W_out = (const float*)d_in[6];
  const float* b_out = (const float*)d_in[7];
  const float* W_ih  = (const float*)d_in[8];
  const float* W_hh  = (const float*)d_in[9];
  const float* b_ih  = (const float*)d_in[10];
  const float* b_hh  = (const float*)d_in[11];
  const float* W_h2h = (const float*)d_in[12];
  const float* b_h2h = (const float*)d_in[13];
  // d_in[14] is H; graph capture needs a fixed launch count -> HSTEPS=24 compiled in.

  float* A     = (float*)d_out;                             // (B,H,NTAR,HDIM)
  float* attnW = A + (size_t)BATCH * HSTEPS * NTAR * HDIM;  // (B,NTAR,NOBS)

  // ---- workspace carve-up (f16 region first, then f32; ~41 MB total) ----
  _Float16* fp = (_Float16*)d_ws;
  _Float16* phi16   = fp; fp += (size_t)NTAR * MPAD;        // 512x256
  _Float16* Wphi16  = fp; fp += (size_t)HDIM * MPAD;        // 128x256
  _Float16* Win16   = fp; fp += (size_t)3 * HDIM * HDIM;    // 384x128
  _Float16* Wout16  = fp; fp += (size_t)HDIM * HDIM;
  _Float16* Wh2h16  = fp; fp += (size_t)HDIM * HDIM;
  _Float16* Wih16   = fp; fp += (size_t)3 * HDIM * HDIM;
  _Float16* Whh16   = fp; fp += (size_t)3 * HDIM * HDIM;
  _Float16* Hobs16  = fp; fp += (size_t)BATCH * NOBS * HDIM;
  _Float16* query16 = fp; fp += (size_t)NTAR * HDIM;
  _Float16* qbuf16  = fp; fp += (size_t)NTAR * HDIM;
  _Float16* kbuf16  = fp; fp += (size_t)BATCH * NOBS * HDIM;
  _Float16* vbufT   = fp; fp += (size_t)BATCH * HDIM * NOBS;
  _Float16* attn16  = fp; fp += (size_t)BATCH * NTAR * HDIM;
  _Float16* h16a    = fp; fp += (size_t)BATCH * NTAR * HDIM;
  _Float16* h16b    = fp; fp += (size_t)BATCH * NTAR * HDIM;
  float* gp = (float*)fp;
  float* h32a = gp; gp += (size_t)BATCH * NTAR * HDIM;
  float* h32b = gp; gp += (size_t)BATCH * NTAR * HDIM;

  dim3 blk(32, 1, 1);  // one wave32 per block; EXEC all-ones for WMMA

  // ---- stage operands to f16 ----
  pad_cvt_f16<<<(NTAR * MPAD + 255) / 256, 256, 0, stream>>>(phi_t, phi16, NTAR, MDIM, MPAD);
  pad_cvt_f16<<<(HDIM * MPAD + 255) / 256, 256, 0, stream>>>(W_phi, Wphi16, HDIM, MDIM, MPAD);
  cvt_f16<<<(3 * HDIM * HDIM + 255) / 256, 256, 0, stream>>>(W_in, Win16, 3 * HDIM * HDIM);
  cvt_f16<<<(HDIM * HDIM + 255) / 256, 256, 0, stream>>>(W_out, Wout16, HDIM * HDIM);
  cvt_f16<<<(HDIM * HDIM + 255) / 256, 256, 0, stream>>>(W_h2h, Wh2h16, HDIM * HDIM);
  cvt_f16<<<(3 * HDIM * HDIM + 255) / 256, 256, 0, stream>>>(W_ih, Wih16, 3 * HDIM * HDIM);
  cvt_f16<<<(3 * HDIM * HDIM + 255) / 256, 256, 0, stream>>>(W_hh, Whh16, 3 * HDIM * HDIM);
  cvt_f16<<<(BATCH * NOBS * HDIM + 255) / 256, 256, 0, stream>>>(H_obs, Hobs16,
                                                                 BATCH * NOBS * HDIM);

  // ---- projections ----
  gemm_f16<1><<<dim3(NTAR / 16, HDIM / 64), blk, 0, stream>>>(
      phi16, MPAD, Wphi16, MPAD, b_phi, query16, HDIM, MPAD);
  gemm_f16<1><<<dim3(NTAR / 16, HDIM / 64), blk, 0, stream>>>(
      query16, HDIM, Win16, HDIM, b_in, qbuf16, HDIM, HDIM);
  gemm_f16<1><<<dim3(BATCH * NOBS / 16, HDIM / 64), blk, 0, stream>>>(
      Hobs16, HDIM, Win16 + (size_t)HDIM * HDIM, HDIM, b_in + HDIM, kbuf16, HDIM, HDIM);
  gemm_f16<2><<<dim3(BATCH * NOBS / 16, HDIM / 64), blk, 0, stream>>>(
      Hobs16, HDIM, Win16 + (size_t)2 * HDIM * HDIM, HDIM, b_in + 2 * HDIM, vbufT,
      NOBS, HDIM);

  // ---- attention (+ head-mean attn_weights straight into d_out) ----
  attn_kernel<<<dim3(NTAR / 16, BATCH), blk, 0, stream>>>(qbuf16, kbuf16, vbufT,
                                                          attn16, attnW);

  // ---- h0 = attn_out @ W_out^T + b_out ----
  gemm_f16<0><<<dim3(BATCH * NTAR / 16, HDIM / 64), blk, 0, stream>>>(
      attn16, HDIM, Wout16, HDIM, b_out, h32a, HDIM, HDIM);
  seed_state<<<(BATCH * NTAR * HDIM + 255) / 256, 256, 0, stream>>>(h32a, A, h16a);

  // ---- GRU rollout: 23 fused steps, 4-wave blocks, weights in LDS ----
  const size_t gruLds = (size_t)(114688 + 4 * 16 * HDIM) * sizeof(_Float16);  // 240 KB
  float* h32in = h32a; float* h32out = h32b;
  _Float16* h16in = h16a; _Float16* h16out = h16b;
  for (int t = 1; t < HSTEPS; ++t) {
    gru_step<<<dim3(BATCH * NTAR / 64), dim3(128), gruLds, stream>>>(
        h32in, h16in, h32out, h16out, Wh2h16, b_h2h, Wih16, b_ih, Whh16, b_hh, A, t);
    float* t32 = h32in; h32in = h32out; h32out = t32;
    _Float16* t16 = h16in; h16in = h16out; h16out = t16;
  }

  (void)in_sizes; (void)n_in; (void)out_size; (void)ws_size;
}
